// Correlation_53644141527260
// MI455X (gfx1250) — matrix operand order repaired and checked by reading, not compiled, not run on hardware
//
#include <hip/hip_runtime.h>

// ---------------------------------------------------------------------------
// PWC-Net cost volume on gfx1250 via bf16 WMMA (Toeplitz/2-N-tile trick).
// out[b,(dy+4)*9+(dx+4),y,x] = (1/C) sum_c first[b,c,y,x]*second[b,c,y+dy,x+dx]
//
// Roofline: 353 MB min HBM traffic -> ~15us at 23.3 TB/s; WMMA work (incl.
// 72% Toeplitz waste) is ~19 GFLOP, far below the memory floor at bf16 WMMA
// rates => memory-bound design. first/out are single-touch (non-temporal);
// second has 9x cross-y reuse kept in L2 (regular temporal).
// ---------------------------------------------------------------------------

typedef __attribute__((ext_vector_type(16))) __bf16          v16bf;
typedef __attribute__((ext_vector_type(8)))  float           v8f;
typedef __attribute__((ext_vector_type(4)))  float           v4f;
typedef __attribute__((ext_vector_type(16))) unsigned short  v16u;
typedef __attribute__((ext_vector_type(8)))  unsigned short  v8u;

#define CB   8      // batch
#define CC   128    // channels (K dimension)
#define CH   128    // height
#define CW   256    // width
#define PAD  4      // max displacement
#define ND   9      // 9 displacements per axis
#define XB   128    // x-pixels per workgroup (8 waves x 16-pixel M-tiles)
#define AST  136    // LDS row stride in elements (mult of 8, avoids 64-bank conflicts)
#define QN   136    // staged q-positions per row (covers x0-4..x0+19 for all 8 tiles)

// LDS plan (dynamic, aliased):  phase 1: sA[XB][AST] bf16 (34816 B)
//                               phase 2: sB[QN][AST] bf16 (36992 B) + sC[8][2][32][8] f32 (16384 B)
#define SB_BYTES   (QN * AST * 2)
#define SC_BYTES   (8 * 2 * 32 * 8 * 4)
#define SMEM_BYTES (SB_BYTES + SC_BYTES)   // 53376 < 64KB -> ~6 workgroups/WGP

__device__ __forceinline__ unsigned short f2bf(float f) {
    unsigned u = __builtin_bit_cast(unsigned, f);
    u += 0x7FFFu + ((u >> 16) & 1u);       // round-to-nearest-even
    return (unsigned short)(u >> 16);
}

__device__ __forceinline__ v16bf ld_frag(const unsigned short* p0, const unsigned short* p1) {
    v8u lo = *(const v8u*)p0;              // ds_load_b128
    v8u hi = *(const v8u*)p1;              // ds_load_b128
    v16u v = __builtin_shufflevector(lo, hi, 0,1,2,3,4,5,6,7,8,9,10,11,12,13,14,15);
    return __builtin_bit_cast(v16bf, v);
}

__global__ __launch_bounds__(256) void corr_wmma_kernel(const float* __restrict__ first,
                                                        const float* __restrict__ second,
                                                        float* __restrict__ out) {
    extern __shared__ unsigned char smem[];
    unsigned short* sA = (unsigned short*)smem;                 // phase 1 only
    unsigned short* sB = (unsigned short*)smem;                 // aliases sA after A-frags preloaded
    float*          sC = (float*)(smem + SB_BYTES);

    const int tid    = threadIdx.x;
    const int wv     = tid >> 5;            // wave id 0..7 -> 16-pixel M-tile
    const int lane   = tid & 31;
    const int half   = (lane >> 4) & 1;     // which K half-chunk this lane owns
    const int ln16   = lane & 15;
    const int xblock = blockIdx.x * XB;
    const int y      = blockIdx.y;
    const int b      = blockIdx.z;
    const int x0     = xblock + 16 * wv;

    // ---- phase 1: stage first[b,:,y,xblock..+127] as bf16 [x][c] ------------
    // float4 global loads (single-touch -> non-temporal), 4x ds_store_b16.
    for (int idx = tid; idx < (XB / 4) * CC; idx += 256) {
        int x = (idx & 31) * 4;
        int c = idx >> 5;
        const float* p = first + (((size_t)b * CC + c) * CH + y) * CW + (xblock + x);
        v4f v = __builtin_nontemporal_load((const v4f*)p);
        #pragma unroll
        for (int j = 0; j < 4; ++j) sA[(x + j) * AST + c] = f2bf(v[j]);
    }
    __syncthreads();

    // ---- preload this wave's A fragments for all 4 K-chunks (K=128) ---------
    // ISA 16-bit A 16x32 layout: lane l<16 row M=l holds K=kc+0..7 (v0-3) and
    // K=kc+16..23 (v4-7); lane l+16 same row holds K=kc+8..15 / kc+24..31.
    v16bf afrag[4];
    {
        const unsigned short* ap = &sA[(16 * wv + ln16) * AST];
        #pragma unroll
        for (int kc = 0; kc < 4; ++kc) {
            const unsigned short* p = ap + kc * 32 + half * 8;
            afrag[kc] = ld_frag(p, p + 16);
        }
    }
    __syncthreads();   // sA dead; sB may now overwrite it

    const float scale = 1.0f / (float)CC;
    float* cw = sC + wv * 512;              // per-wave 2x16x16 f32 scratch

    for (int dyi = 0; dyi < ND; ++dyi) {
        const int yy = y + dyi - PAD;
        const bool rowok = (yy >= 0) && (yy < CH);

        // ---- stage second[b,:,yy, xblock-4 .. xblock+131] as bf16 [q][c] ----
        // zero-filled outside the image => matches reference zero padding.
        // Regular temporal loads: rows are reused 9x across neighboring y WGs.
        for (int idx = tid; idx < (QN / 4) * CC; idx += 256) {
            int q  = (idx % 34) * 4;
            int c  = idx / 34;
            int xg = xblock - PAD + q;      // multiple of 4 -> 16B aligned
            v4f v = {0.0f, 0.0f, 0.0f, 0.0f};
            if (rowok) {
                const float* p = second + (((size_t)b * CC + c) * CH + yy) * CW + xg;
                if (xg >= 0 && xg <= CW - 4) {
                    v = *(const v4f*)p;
                } else {
                    #pragma unroll
                    for (int j = 0; j < 4; ++j)
                        if (xg + j >= 0 && xg + j < CW) v[j] = p[j];
                }
            }
            #pragma unroll
            for (int j = 0; j < 4; ++j) sB[(q + j) * AST + c] = f2bf(v[j]);
        }
        __syncthreads();

        // ---- preload ALL B fragments, then run the 8-WMMA chain -------------
        // B 32x16 layout: lane l<16 col N=l holds K=kc+0..7,kc+16..23; lane l+16 K=kc+8..15,kc+24..31
        v16bf bfrag[2][4];
        {
            const unsigned short* bbase = &sB[(16 * wv + ln16) * AST + half * 8];
            #pragma unroll
            for (int kc = 0; kc < 4; ++kc) {
                const unsigned short* bp0 = bbase + kc * 32;
                const unsigned short* bp1 = bp0 + 16 * AST;
                bfrag[0][kc] = ld_frag(bp0, bp0 + 16);
                bfrag[1][kc] = ld_frag(bp1, bp1 + 16);
            }
        }
        // acc_t[m][n] = sum_c first[x0+m,c] * second[yy, x0-4+16t+n, c]
        v8f acc0 = {}, acc1 = {};
        #pragma unroll
        for (int kc = 0; kc < 4; ++kc) {
            acc0 = __builtin_amdgcn_wmma_f32_16x16x32_bf16(false, afrag[kc], false, bfrag[0][kc],
                                                           (short)0, acc0, false, false);
            acc1 = __builtin_amdgcn_wmma_f32_16x16x32_bf16(false, afrag[kc], false, bfrag[1][kc],
                                                           (short)0, acc1, false, false);
        }

        // ---- dump accumulators: element (m,n) lives at lane n+16*(m>=8), vgpr m&7
        *(v8f*)(cw + lane * 8)       = acc0;
        *(v8f*)(cw + 256 + lane * 8) = acc1;

        __syncthreads();   // all waves done reading sB; sC stores visible

        // ---- scatter 9 diagonals x 16 pixels, scaled by 1/C (streaming store)
        for (int i = lane; i < ND * 16; i += 32) {
            int dxi = i >> 4;               // dx+4 in 0..8
            int m   = i & 15;
            int s   = m + dxi;              // = m + dx + 4, in [0,23]
            int t   = s >> 4;               // which N-tile
            int n   = s & 15;
            float v = cw[t * 256 + (n + ((m & 8) << 1)) * 8 + (m & 7)] * scale;
            size_t o = (((size_t)(b * 81 + dyi * ND + dxi)) * CH + y) * CW + (x0 + m);
            __builtin_nontemporal_store(v, &out[o]);
        }
        // next iteration's sB staging is safe: the barrier above ordered it
        // after all sB reads; scatter only touches sC/global.
    }
}

extern "C" void kernel_launch(void* const* d_in, const int* in_sizes, int n_in,
                              void* d_out, int out_size, void* d_ws, size_t ws_size,
                              hipStream_t stream) {
    (void)in_sizes; (void)n_in; (void)d_ws; (void)ws_size; (void)out_size;
    const float* first  = (const float*)d_in[0];
    const float* second = (const float*)d_in[1];
    float*       out    = (float*)d_out;
    dim3 grid(CW / XB, CH, CB);   // (2, 128, 8)
    corr_wmma_kernel<<<grid, 256, SMEM_BYTES, stream>>>(first, second, out);
}
